// MultiHeadAttention_50792283242934
// MI455X (gfx1250) — matrix operand order repaired and checked by reading, not compiled, not run on hardware
//
#include <hip/hip_runtime.h>
#include <hip/hip_bf16.h>

// Problem constants (from reference)
#define B_    2
#define S_    2048
#define D_    1024
#define H_    16
#define HD_   64
#define SCALE_ 0.125f   // HEAD_DIM^-0.5
#define STRIP 16        // q-rows per fused-attention block

typedef __attribute__((ext_vector_type(16))) _Float16 v16h;
typedef __attribute__((ext_vector_type(8)))  _Float16 v8h;
typedef __attribute__((ext_vector_type(4)))  _Float16 v4h;
typedef __attribute__((ext_vector_type(8)))  float    v8f;
typedef __attribute__((ext_vector_type(4)))  float    v4f_t;
typedef __attribute__((ext_vector_type(4)))  int      v4i_t;

union V16H { v16h v; _Float16 e[16]; v8h h2[2]; };
union V8F  { v8f  v; float e[8]; };

#if defined(__has_builtin)
#if __has_builtin(__builtin_amdgcn_global_store_async_from_lds_b128)
#define HAVE_ASYNC_LDS 1
#endif
#endif

#ifdef HAVE_ASYNC_LDS
typedef v4i_t __attribute__((address_space(1))) as1_v4i;
typedef v4i_t __attribute__((address_space(3))) as3_v4i;
#endif

// ---------------------------------------------------------------------------
// WMMA fragment loaders (wave32 layouts per CDNA5 ISA 7.12.2)
// A (16x32 f16, row-major src): lane l<16  -> M=l,    K {0..7, 16..23}
//                               lane l>=16 -> M=l-16, K {8..15,24..31}
// B (32x16 f16 from row-major [N,K] src): lane l<16  -> N=l,    K {0..7, 16..23}
//                                         lane l>=16 -> N=l-16, K {8..15,24..31}
// ---------------------------------------------------------------------------
__device__ __forceinline__ v16h load_a_f16(const _Float16* __restrict__ base, int lda) {
  int lane = threadIdx.x & 31;
  int l = lane & 15, hi = lane >> 4;
  const _Float16* p = base + (size_t)l * lda + hi * 8;
  V16H r;
  r.h2[0] = *(const v8h*)(p);
  r.h2[1] = *(const v8h*)(p + 16);
  return r.v;
}

__device__ __forceinline__ v16h load_b_f16(const _Float16* __restrict__ base, int ldb) {
  int lane = threadIdx.x & 31;
  int l = lane & 15, hi = lane >> 4;
  const _Float16* p = base + (size_t)l * ldb + hi * 8;
  V16H r;
  r.h2[0] = *(const v8h*)(p);
  r.h2[1] = *(const v8h*)(p + 16);
  return r.v;
}

__device__ __forceinline__ v8f wmma_f16(v16h a, v16h b, v8f c) {
  return __builtin_amdgcn_wmma_f32_16x16x32_f16(false, a, false, b, (short)0, c, false, false);
}

// ---------------------------------------------------------------------------
// fp32 -> f16 convert (vectorized x4)
// ---------------------------------------------------------------------------
__global__ __launch_bounds__(256) void cvt_f32_f16_kernel(
    const float* __restrict__ in, _Float16* __restrict__ out, long n4) {
  long i = (long)blockIdx.x * blockDim.x + threadIdx.x;
  if (i >= n4) return;
  v4f_t f = ((const v4f_t*)in)[i];
  ((v4h*)out)[i] = __builtin_convertvector(f, v4h);
}

// ---------------------------------------------------------------------------
// Generic GEMM with bias: Y[M,N] = A[M,K](f16) * W[N,K](f16)^T + bias[N]
// Block tile 128(M) x 64(N), 8 waves, each wave 32x32 (2x2 WMMA tiles).
// MODE 0: f16 head-split [B,H,S,Hd]; MODE 1: f16 head-transp [B,H,Hd,S];
// MODE 2: fp32 row-major [M,N]
// ---------------------------------------------------------------------------
template <int MODE>
__global__ __launch_bounds__(256) void gemm_bias_kernel(
    const _Float16* __restrict__ A, const _Float16* __restrict__ W,
    const float* __restrict__ bias, void* __restrict__ out,
    int M, int N, int K) {
  int wid = threadIdx.x >> 5;
  int wm = wid >> 1, wn = wid & 1;
  int m0 = blockIdx.y * 128 + wm * 32;
  int n0 = blockIdx.x * 64 + wn * 32;

  v8f acc00 = {}, acc01 = {}, acc10 = {}, acc11 = {};
  #pragma unroll 4
  for (int k0 = 0; k0 < K; k0 += 32) {
    // speculative prefetch of a future K-slice (global_prefetch_b8)
    __builtin_prefetch(A + (size_t)m0 * K + k0 + 128, 0, 1);
    __builtin_prefetch(W + (size_t)n0 * K + k0 + 128, 0, 1);
    v16h a0 = load_a_f16(A + (size_t)m0 * K + k0, K);
    v16h a1 = load_a_f16(A + (size_t)(m0 + 16) * K + k0, K);
    v16h b0 = load_b_f16(W + (size_t)n0 * K + k0, K);
    v16h b1 = load_b_f16(W + (size_t)(n0 + 16) * K + k0, K);
    acc00 = wmma_f16(a0, b0, acc00);
    acc01 = wmma_f16(a0, b1, acc01);
    acc10 = wmma_f16(a1, b0, acc10);
    acc11 = wmma_f16(a1, b1, acc11);
  }

  int lane = threadIdx.x & 31;
  int l = lane & 15, hi = lane >> 4;
  V8F av[2][2];
  av[0][0].v = acc00; av[0][1].v = acc01; av[1][0].v = acc10; av[1][1].v = acc11;

  #pragma unroll
  for (int i = 0; i < 2; ++i)
    #pragma unroll
    for (int j = 0; j < 2; ++j) {
      int n = n0 + j * 16 + l;
      float bv = bias[n];
      #pragma unroll
      for (int r = 0; r < 8; ++r) {
        int m = m0 + i * 16 + hi * 8 + r;
        float val = av[i][j].e[r] + bv;
        if (MODE == 2) {
          ((float*)out)[(size_t)m * N + n] = val;
        } else {
          int b = m / S_, s = m % S_;
          int h = n / HD_, hd = n % HD_;
          size_t idx;
          if (MODE == 0) idx = (((size_t)(b * H_ + h)) * S_ + s) * HD_ + hd;
          else           idx = (((size_t)(b * H_ + h)) * HD_ + hd) * S_ + s;
          ((_Float16*)out)[idx] = (_Float16)val;
        }
      }
    }
}

// ---------------------------------------------------------------------------
// Fused attention: per (z = b*H + h, strip of 16 q-rows)
//   1) scores(16 x 2048) = scale * Q K^T, +mask          -> LDS
//   2) row softmax over LDS
//   3) write probs fp32 -> d_out (async LDS->global when available)
//   4) PV: out(16 x 64) = probs @ V  (A-frags from LDS, cross-wave ds_add_f32)
//   5) store f16 attn output tile
// LDS: 16*(2048+4)*4 + 16*64*4 + 16*16*4 ~= 136.5 KB -> 2 blocks/WGP (320 KB)
// ---------------------------------------------------------------------------
__global__ __launch_bounds__(256) void fused_attn_kernel(
    const _Float16* __restrict__ qh,   // [B*H][S][Hd]
    const _Float16* __restrict__ kh,   // [B*H][S][Hd]
    const _Float16* __restrict__ vT,   // [B*H][Hd][S]
    const int* __restrict__ mask,      // [B][S][S]
    float* __restrict__ probs_out,     // [B*H][S][S]  (attn_weights region of d_out)
    _Float16* __restrict__ attnOut) {  // [B][S][D] f16
  __shared__ float probs[STRIP][S_ + 4];   // +4 pad: A-frag LDS reads spread banks
  __shared__ float obuf[STRIP][HD_];
  __shared__ float red[STRIP][16];

  const int z = blockIdx.y;
  const int m0g = blockIdx.x * STRIP;
  const _Float16* Q  = qh + ((size_t)z * S_ + m0g) * HD_;
  const _Float16* Kk = kh + (size_t)z * S_ * HD_;
  const _Float16* Vt = vT + (size_t)z * HD_ * S_;
  const int* mk = mask + (size_t)(z / H_) * S_ * S_ + (size_t)m0g * S_;
  float* pout = probs_out + ((size_t)z * S_ + m0g) * S_;

  const int tid = threadIdx.x;
  const int wid = tid >> 5;
  const int lane = tid & 31;
  const int l = lane & 15, hi = lane >> 4;

  // zero PV accumulation buffer (visible after first barrier)
  for (int i = tid; i < STRIP * HD_; i += 256) (&obuf[0][0])[i] = 0.f;

  // ---- Phase 1: scores -> LDS --------------------------------------------
  v16h qa0 = load_a_f16(Q, HD_);        // K 0..31
  v16h qa1 = load_a_f16(Q + 32, HD_);   // K 32..63
  for (int j = wid * 32; j < S_; j += 256) {
    v16h b00 = load_b_f16(Kk + (size_t)j * HD_, HD_);
    v16h b01 = load_b_f16(Kk + (size_t)j * HD_ + 32, HD_);
    v16h b10 = load_b_f16(Kk + (size_t)(j + 16) * HD_, HD_);
    v16h b11 = load_b_f16(Kk + (size_t)(j + 16) * HD_ + 32, HD_);
    v8f acc0 = {}, acc1 = {};
    acc0 = wmma_f16(qa0, b00, acc0);
    acc0 = wmma_f16(qa1, b01, acc0);
    acc1 = wmma_f16(qa0, b10, acc1);
    acc1 = wmma_f16(qa1, b11, acc1);
    V8F a0, a1; a0.v = acc0; a1.v = acc1;
    #pragma unroll
    for (int t2 = 0; t2 < 2; ++t2) {
      int n = j + t2 * 16 + l;
      const V8F& av = t2 ? a1 : a0;
      #pragma unroll
      for (int r = 0; r < 8; ++r) {
        int m = hi * 8 + r;
        float val = av.e[r] * SCALE_;
        if (mk[(size_t)m * S_ + n] == 0) val = -1e9f;
        probs[m][n] = val;
      }
    }
  }
  __syncthreads();

  // ---- Phase 2: softmax over LDS rows ------------------------------------
  {
    int row = tid >> 4, seg = tid & 15;
    float* pr = probs[row];
    float mx = -3.4e38f;
    #pragma unroll 8
    for (int i = 0; i < 128; ++i) mx = fmaxf(mx, pr[seg + i * 16]);
    red[row][seg] = mx;
    __syncthreads();
    mx = red[row][0];
    #pragma unroll
    for (int i = 1; i < 16; ++i) mx = fmaxf(mx, red[row][i]);
    __syncthreads();
    float sum = 0.f;
    #pragma unroll 8
    for (int i = 0; i < 128; ++i) {
      int c = seg + i * 16;
      float e = __expf(pr[c] - mx);
      pr[c] = e;
      sum += e;
    }
    red[row][seg] = sum;
    __syncthreads();
    sum = 0.f;
    #pragma unroll
    for (int i = 0; i < 16; ++i) sum += red[row][i];
    float inv = 1.0f / sum;
    #pragma unroll 8
    for (int i = 0; i < 128; ++i) pr[seg + i * 16] *= inv;
  }
  __syncthreads();

  // ---- Phase 3: write fp32 probs to d_out --------------------------------
  {
    int row = tid >> 4, seg = tid & 15;
    #pragma unroll 4
    for (int it = 0; it < 32; ++it) {
      int col = seg * 4 + it * 64;
#ifdef HAVE_ASYNC_LDS
      __builtin_amdgcn_global_store_async_from_lds_b128(
          (as1_v4i*)(pout + (size_t)row * S_ + col),
          (as3_v4i*)&probs[row][col], 0, 0);
#else
      *(v4f_t*)(pout + (size_t)row * S_ + col) = *(const v4f_t*)&probs[row][col];
#endif
    }
  }
  // no wait needed before PV: async stores only READ probs; S_ENDPGM drains ASYNCcnt.

  // ---- Phase 4: PV from LDS probs (each wave owns a 256-wide K chunk) ----
  {
    int k0w = wid * 256;
    v8f o0 = {}, o1 = {}, o2 = {}, o3 = {};
    for (int kk = 0; kk < 256; kk += 32) {
      int k0 = k0w + kk;
      const float* pa = &probs[l][k0 + hi * 8];
      v4f_t f0 = *(const v4f_t*)(pa);
      v4f_t f1 = *(const v4f_t*)(pa + 4);
      v4f_t f2 = *(const v4f_t*)(pa + 16);
      v4f_t f3 = *(const v4f_t*)(pa + 20);
      V16H a;
      *(v4h*)&a.e[0]  = __builtin_convertvector(f0, v4h);
      *(v4h*)&a.e[4]  = __builtin_convertvector(f1, v4h);
      *(v4h*)&a.e[8]  = __builtin_convertvector(f2, v4h);
      *(v4h*)&a.e[12] = __builtin_convertvector(f3, v4h);
      v16h b0 = load_b_f16(Vt + k0, S_);
      v16h b1 = load_b_f16(Vt + (size_t)16 * S_ + k0, S_);
      v16h b2 = load_b_f16(Vt + (size_t)32 * S_ + k0, S_);
      v16h b3 = load_b_f16(Vt + (size_t)48 * S_ + k0, S_);
      o0 = wmma_f16(a.v, b0, o0);
      o1 = wmma_f16(a.v, b1, o1);
      o2 = wmma_f16(a.v, b2, o2);
      o3 = wmma_f16(a.v, b3, o3);
    }
    V8F t[4]; t[0].v = o0; t[1].v = o1; t[2].v = o2; t[3].v = o3;
    #pragma unroll
    for (int tt = 0; tt < 4; ++tt)
      #pragma unroll
      for (int r = 0; r < 8; ++r)
        atomicAdd(&obuf[hi * 8 + r][tt * 16 + l], t[tt].e[r]);  // ds_add_f32
  }
  __syncthreads();

  // ---- Phase 5: store f16 attn-output tile -------------------------------
  {
    int idx = tid * 4;                 // 16*64 / 256 threads = 4 each
    int m = idx >> 6, c = idx & 63;
    int b = z / H_, h = z % H_;
    v4f_t f = *(const v4f_t*)&obuf[m][c];
    size_t o = ((size_t)(b * S_ + m0g + m)) * D_ + h * HD_ + c;
    *(v4h*)(attnOut + o) = __builtin_convertvector(f, v4h);
  }
}

// ---------------------------------------------------------------------------
extern "C" void kernel_launch(void* const* d_in, const int* in_sizes, int n_in,
                              void* d_out, int out_size, void* d_ws, size_t ws_size,
                              hipStream_t stream) {
  const float* query = (const float*)d_in[0];
  const float* key   = (const float*)d_in[1];
  const float* value = (const float*)d_in[2];
  const int*   mask  = (const int*)d_in[3];
  const float* Wq = (const float*)d_in[4];
  const float* bq = (const float*)d_in[5];
  const float* Wk = (const float*)d_in[6];
  const float* bk = (const float*)d_in[7];
  const float* Wv = (const float*)d_in[8];
  const float* bv = (const float*)d_in[9];
  const float* Wo = (const float*)d_in[10];
  const float* bo = (const float*)d_in[11];

  float* out_proj = (float*)d_out;                       // [B,S,D]
  float* attnW    = out_proj + (size_t)B_ * S_ * D_;     // [B,H,S,S]

  const size_t XN = (size_t)B_ * S_ * D_;  // 4M elements
  const size_t WN = (size_t)D_ * D_;       // 1M elements
  _Float16* ws    = (_Float16*)d_ws;
  _Float16* hq_in = ws;
  _Float16* hk_in = hq_in + XN;
  _Float16* hv_in = hk_in + XN;
  _Float16* hWq   = hv_in + XN;
  _Float16* hWk   = hWq + WN;
  _Float16* hWv   = hWk + WN;
  _Float16* hWo   = hWv + WN;
  _Float16* qh    = hWo + WN;              // q heads   [B,H,S,Hd]
  _Float16* kh    = qh + XN;               // k heads   [B,H,S,Hd]
  _Float16* vT    = kh + XN;               // v heads^T [B,H,Hd,S]
  _Float16* aOut  = vT + XN;               // attn out  [B,S,D] f16

  dim3 blk(256);

  {
    long n4 = (long)(XN / 4);
    dim3 g((unsigned)((n4 + 255) / 256));
    cvt_f32_f16_kernel<<<g, blk, 0, stream>>>(query, hq_in, n4);
    cvt_f32_f16_kernel<<<g, blk, 0, stream>>>(key,   hk_in, n4);
    cvt_f32_f16_kernel<<<g, blk, 0, stream>>>(value, hv_in, n4);
  }
  {
    long n4 = (long)(WN / 4);
    dim3 g((unsigned)((n4 + 255) / 256));
    cvt_f32_f16_kernel<<<g, blk, 0, stream>>>(Wq, hWq, n4);
    cvt_f32_f16_kernel<<<g, blk, 0, stream>>>(Wk, hWk, n4);
    cvt_f32_f16_kernel<<<g, blk, 0, stream>>>(Wv, hWv, n4);
    cvt_f32_f16_kernel<<<g, blk, 0, stream>>>(Wo, hWo, n4);
  }

  const int M = B_ * S_, N = D_, K = D_;
  dim3 gproj(N / 64, M / 128, 1);
  gemm_bias_kernel<0><<<gproj, blk, 0, stream>>>(hq_in, hWq, bq, qh, M, N, K);
  gemm_bias_kernel<0><<<gproj, blk, 0, stream>>>(hk_in, hWk, bk, kh, M, N, K);
  gemm_bias_kernel<1><<<gproj, blk, 0, stream>>>(hv_in, hWv, bv, vT, M, N, K);

  dim3 gattn(S_ / STRIP, B_ * H_, 1);
  fused_attn_kernel<<<gattn, blk, 0, stream>>>(qh, kh, vT, mask, attnW, aOut);

  gemm_bias_kernel<2><<<gproj, blk, 0, stream>>>(aOut, hWo, bo, out_proj, M, N, K);
}
